// _NonLocalBlockND_16965120820002
// MI455X (gfx1250) — compile-verified, matrix operand order checked by reading, not run on hardware
//
#include <hip/hip_runtime.h>

#define BATCH 32
#define CH    256
#define ICH   128
#define NPIX  1024
#define PROJW 256   // columns: theta[0:128) | phi[128:256)  (g stored transposed in GT)

#define XSTR  264   // x-tile LDS row stride (ushorts): 256 + 8 pad -> 4-dword lane skew
#define PBSTR 1032  // prob LDS row stride: 1024 + 8 pad
#define YSTR  136   // y LDS row stride: 128 + 8 pad

typedef __attribute__((ext_vector_type(16))) __bf16 v16bf;
typedef __attribute__((ext_vector_type(8)))  float  v8f;

union Frag { v16bf v; unsigned short u[16]; };

__device__ __forceinline__ unsigned short f2bf(float f) {
  unsigned int u = __float_as_uint(f);
  u += 0x7fffu + ((u >> 16) & 1u);      // round-to-nearest-even
  return (unsigned short)(u >> 16);
}

// K index held by element i of a 16-bit A/B fragment for lane `lane`
// (ISA 7.12.2: lanes 0-15 hold K 0..7,16..23; lanes 16-31 hold K 8..15,24..31)
__device__ __forceinline__ int kmap(int i, int lane) {
  return ((i >> 3) << 4) + ((lane >> 4) << 3) + (i & 7);
}

__device__ __forceinline__ v8f wmma_bf16(const Frag& a, const Frag& b, v8f c) {
  return __builtin_amdgcn_wmma_f32_16x16x32_bf16(false, a.v, false, b.v,
                                                 (short)0, c, false, false);
}

__device__ __forceinline__ v8f vzero8() {
  v8f z = {0.f, 0.f, 0.f, 0.f, 0.f, 0.f, 0.f, 0.f};
  return z;
}

// ---------------------------------------------------------------------------
// Pack projection weights into bf16 with the layouts the WMMA kernels want.
// ---------------------------------------------------------------------------
__global__ void nlb_prep(const float* __restrict__ Wg, const float* __restrict__ bg,
                         const float* __restrict__ Wth, const float* __restrict__ bth,
                         const float* __restrict__ Wph, const float* __restrict__ bph,
                         const float* __restrict__ Ww,
                         unsigned short* __restrict__ Wall,
                         unsigned short* __restrict__ WwB,
                         float* __restrict__ bias_all) {
  const int idx = blockIdx.x * blockDim.x + threadIdx.x;
  if (idx < 384 * CH) {
    const int o = idx / CH, c = idx % CH;
    const float w = (o < ICH)     ? Wth[o * CH + c]
                  : (o < 2 * ICH) ? Wph[(o - ICH) * CH + c]
                                  : Wg[(o - 2 * ICH) * CH + c];
    Wall[idx] = f2bf(w);
  } else if (idx < 384 * CH + CH * ICH) {
    const int j = idx - 384 * CH;
    WwB[j] = f2bf(Ww[j]);
  } else if (idx < 384 * CH + CH * ICH + 384) {
    const int o = idx - 384 * CH - CH * ICH;
    bias_all[o] = (o < ICH) ? bth[o] : (o < 2 * ICH) ? bph[o - ICH] : bg[o - 2 * ICH];
  }
}

// ---------------------------------------------------------------------------
// Projections. One block = (b, 16 pixels of N). GEMM: xT(16xC) @ WallT(Cx384).
// theta|phi -> PROJ[b][n][0:256) (row-major per pixel)
// g         -> GT[b][ic][n]      (transposed: contiguous-k for phase-3 B frags)
// ---------------------------------------------------------------------------
__global__ void __launch_bounds__(256) nlb_proj(const float* __restrict__ x,
                                                const unsigned short* __restrict__ Wall,
                                                const float* __restrict__ bias_all,
                                                unsigned short* __restrict__ PROJ,
                                                unsigned short* __restrict__ GT) {
  __shared__ unsigned short xs[16 * XSTR];  // 8.25 KB: x tile, [n][c] bf16 (padded)
  const int b   = blockIdx.x >> 6;
  const int n0  = (blockIdx.x & 63) * 16;
  const int tid = threadIdx.x;

  const float* xb = x + (size_t)b * CH * NPIX;
  for (int i = tid; i < CH * 16; i += 256) {
    const int c = i >> 4, j = i & 15;          // coalesced read, [n][c] LDS write
    xs[j * XSTR + c] = f2bf(xb[(size_t)c * NPIX + n0 + j]);
  }
  __syncthreads();

  const int wave = tid >> 5, lane = tid & 31;
  const int mrow = lane & 15, half = lane >> 4;

  v8f acc[3] = {vzero8(), vzero8(), vzero8()};   // each wave: 3 of 24 col tiles
  for (int kk = 0; kk < 8; ++kk) {               // K = 256
    const int k0 = kk * 32;
    Frag a;
#pragma unroll
    for (int i = 0; i < 16; ++i)                 // contiguous-k -> ds_load_b128 x2
      a.u[i] = xs[mrow * XSTR + k0 + kmap(i, lane)];
#pragma unroll
    for (int t = 0; t < 3; ++t) {
      const int ocol = (wave * 3 + t) * 16 + mrow;
      Frag bm;
#pragma unroll
      for (int i = 0; i < 16; ++i)               // contiguous -> global_load_b128 x2
        bm.u[i] = Wall[ocol * CH + k0 + kmap(i, lane)];
      acc[t] = wmma_bf16(a, bm, acc[t]);
    }
  }

  unsigned short* Po  = PROJ + (size_t)b * NPIX * PROJW;
  unsigned short* GTb = GT + (size_t)b * ICH * NPIX;
#pragma unroll
  for (int t = 0; t < 3; ++t) {
    const int tile = wave * 3 + t;               // uniform per wave
    const int ocol = tile * 16 + mrow;
    const float bias = bias_all[ocol];
    if (tile < 16) {                             // theta | phi, [n][o] layout
#pragma unroll
      for (int r = 0; r < 8; ++r)
        Po[(size_t)(n0 + half * 8 + r) * PROJW + ocol] = f2bf(acc[t][r] + bias);
    } else {                                     // g, transposed [ic][n], b128 store
      const int ic = ocol - 256;
      unsigned int pk[4];
#pragma unroll
      for (int p = 0; p < 4; ++p)
        pk[p] = (unsigned int)f2bf(acc[t][2 * p] + bias) |
                ((unsigned int)f2bf(acc[t][2 * p + 1] + bias) << 16);
      *(uint4*)(GTb + (size_t)ic * NPIX + n0 + half * 8) = *(uint4*)pk;
    }
  }
}

// ---------------------------------------------------------------------------
// Fused attention: one block owns 16 query rows of one batch.
// scores (16x1024) stay in registers; softmax via shuffles + tiny LDS;
// probs -> bf16 LDS; y = P@g; wy = Ww@yT; BN partial sums via atomics.
// ---------------------------------------------------------------------------
__global__ void __launch_bounds__(256) nlb_attn(const unsigned short* __restrict__ PROJ,
                                                const unsigned short* __restrict__ GT,
                                                const unsigned short* __restrict__ WwB,
                                                const float* __restrict__ bw,
                                                float* __restrict__ WY,
                                                float* __restrict__ chsum,
                                                float* __restrict__ chsumsq) {
  __shared__ unsigned short pb[16 * PBSTR];  // 33 KB: probs bf16 (later reused as wred)
  __shared__ unsigned short yb[16 * YSTR];   // 4.25 KB: y tile bf16
  __shared__ float red[16 * 8];
  __shared__ float rowmax[16];
  __shared__ float rowsum[16];

  const int b    = blockIdx.x >> 6;
  const int n0   = (blockIdx.x & 63) * 16;
  const int tid  = threadIdx.x;
  const int wave = tid >> 5, lane = tid & 31;
  const int mrow = lane & 15, half = lane >> 4;

  const unsigned short* Pb  = PROJ + (size_t)b * NPIX * PROJW;
  const unsigned short* GTb = GT + (size_t)b * ICH * NPIX;

  // Warm L2/WGP$ with this batch's g-slice (256 KB, reused by 64 blocks):
  // one 128B line per prefetch -> global_prefetch_b8.
#pragma unroll
  for (int i = 0; i < 8; ++i)
    __builtin_prefetch(GTb + ((size_t)tid * 8 + i) * 64, 0, 3);

  // ---- phase 1: scores; each wave covers 128 key columns, K = IC = 128 ----
  Frag ath[4];
#pragma unroll
  for (int kk = 0; kk < 4; ++kk)
#pragma unroll
    for (int i = 0; i < 16; ++i)
      ath[kk].u[i] = Pb[(size_t)(n0 + mrow) * PROJW + kk * 32 + kmap(i, lane)];

  v8f acc[8];
#pragma unroll
  for (int tt = 0; tt < 8; ++tt) {
    acc[tt] = vzero8();
    const int mcol = (wave * 8 + tt) * 16 + mrow;
#pragma unroll
    for (int kk = 0; kk < 4; ++kk) {
      Frag bm;
#pragma unroll
      for (int i = 0; i < 16; ++i)
        bm.u[i] = Pb[(size_t)mcol * PROJW + ICH + kk * 32 + kmap(i, lane)];
      acc[tt] = wmma_bf16(ath[kk], bm, acc[tt]);
    }
  }

  // ---- phase 2: softmax over 1024 columns (registers + shuffles + LDS) ----
  float lmax[8];
#pragma unroll
  for (int r = 0; r < 8; ++r) {
    float m = acc[0][r];
#pragma unroll
    for (int tt = 1; tt < 8; ++tt) m = fmaxf(m, acc[tt][r]);
#pragma unroll
    for (int s = 1; s < 16; s <<= 1) m = fmaxf(m, __shfl_xor(m, s, 32));
    lmax[r] = m;
  }
  if (mrow == 0)
#pragma unroll
    for (int r = 0; r < 8; ++r) red[(half * 8 + r) * 8 + wave] = lmax[r];
  __syncthreads();
  if (tid < 16) {
    float m = red[tid * 8];
    for (int w = 1; w < 8; ++w) m = fmaxf(m, red[tid * 8 + w]);
    rowmax[tid] = m;
  }
  __syncthreads();

#pragma unroll
  for (int r = 0; r < 8; ++r) {
    const float rm = rowmax[half * 8 + r];
    float s = 0.f;
#pragma unroll
    for (int tt = 0; tt < 8; ++tt) {
      const float e = __expf(acc[tt][r] - rm);
      acc[tt][r] = e;
      s += e;
    }
#pragma unroll
    for (int sh = 1; sh < 16; sh <<= 1) s += __shfl_xor(s, sh, 32);
    if (mrow == 0) red[(half * 8 + r) * 8 + wave] = s;
  }
  __syncthreads();
  if (tid < 16) {
    float s = 0.f;
    for (int w = 0; w < 8; ++w) s += red[tid * 8 + w];
    rowsum[tid] = s;
  }
  __syncthreads();

#pragma unroll
  for (int r = 0; r < 8; ++r) {
    const int row = half * 8 + r;
    const float inv = 1.f / rowsum[row];
#pragma unroll
    for (int tt = 0; tt < 8; ++tt)
      pb[row * PBSTR + wave * 128 + tt * 16 + mrow] = f2bf(acc[tt][r] * inv);
  }
  __syncthreads();

  // ---- phase 3: y[16,128] = P @ g, K = 1024; one ic tile per wave ----
  {
    v8f accy = vzero8();
    const int icol = wave * 16 + mrow;
    for (int kk = 0; kk < 32; ++kk) {
      const int k0 = kk * 32;
      Frag a, bm;
#pragma unroll
      for (int i = 0; i < 16; ++i) {
        const int k = k0 + kmap(i, lane);
        a.u[i]  = pb[mrow * PBSTR + k];          // contiguous -> ds_load_b128 x2
        bm.u[i] = GTb[(size_t)icol * NPIX + k];  // contiguous -> global_load_b128 x2
      }
      accy = wmma_bf16(a, bm, accy);
    }
#pragma unroll
    for (int r = 0; r < 8; ++r)
      yb[(half * 8 + r) * YSTR + icol] = f2bf(accy[r]);
  }
  __syncthreads();

  // ---- phase 4: wy[256,16] = Ww @ yT (+bw); store + BN partial sums ----
  float* wred = (float*)pb;   // reuse 33KB region (needs 16KB)
#pragma unroll
  for (int t = 0; t < 2; ++t) {
    const int crow = (wave * 2 + t) * 16 + mrow;
    v8f accw = vzero8();
#pragma unroll
    for (int kk = 0; kk < 4; ++kk) {    // K = IC = 128
      const int k0 = kk * 32;
      Frag a, bm;
#pragma unroll
      for (int i = 0; i < 16; ++i) {
        const int k = k0 + kmap(i, lane);
        a.u[i]  = WwB[crow * ICH + k];           // contiguous global
        bm.u[i] = yb[mrow * YSTR + k];           // contiguous LDS
      }
      accw = wmma_bf16(a, bm, accw);
    }
#pragma unroll
    for (int r = 0; r < 8; ++r) {
      const int c = (wave * 2 + t) * 16 + half * 8 + r;
      const float v = accw[r] + bw[c];
      WY[((size_t)b * CH + c) * NPIX + n0 + mrow] = v;
      wred[c * 16 + mrow] = v;
    }
  }
  __syncthreads();
  if (tid < CH) {
    float s = 0.f, s2 = 0.f;
#pragma unroll
    for (int j = 0; j < 16; ++j) {
      const float v = wred[tid * 16 + j];
      s += v;
      s2 += v * v;
    }
    atomicAdd(&chsum[tid], s);
    atomicAdd(&chsumsq[tid], s2);
  }
}

// ---------------------------------------------------------------------------
// BN statistics -> per-channel scale/shift.
// ---------------------------------------------------------------------------
__global__ void nlb_stats(const float* __restrict__ chsum, const float* __restrict__ chsumsq,
                          const float* __restrict__ gamma, const float* __restrict__ beta,
                          float* __restrict__ scale, float* __restrict__ shift) {
  const int c = threadIdx.x;
  if (c < CH) {
    const float cnt  = (float)(BATCH * NPIX);
    const float mean = chsum[c] / cnt;
    const float var  = chsumsq[c] / cnt - mean * mean;
    const float sc   = gamma[c] * rsqrtf(var + 1e-5f);
    scale[c] = sc;
    shift[c] = beta[c] - mean * sc;
  }
}

// ---------------------------------------------------------------------------
// z = BN(wy) + x, float4-vectorized.
// ---------------------------------------------------------------------------
__global__ void nlb_apply(const float* __restrict__ WY, const float* __restrict__ x,
                          const float* __restrict__ scale, const float* __restrict__ shift,
                          float* __restrict__ out) {
  const size_t i4 = ((size_t)blockIdx.x * blockDim.x + threadIdx.x) * 4;
  const int c = (int)((i4 >> 10) & (CH - 1));      // layout [b][c][n], N=1024
  const float4 wy = *(const float4*)(WY + i4);
  const float4 xv = *(const float4*)(x + i4);
  const float sc = scale[c], sh = shift[c];
  float4 o;
  o.x = wy.x * sc + sh + xv.x;
  o.y = wy.y * sc + sh + xv.y;
  o.z = wy.z * sc + sh + xv.z;
  o.w = wy.w * sc + sh + xv.w;
  *(float4*)(out + i4) = o;
}

// ---------------------------------------------------------------------------
extern "C" void kernel_launch(void* const* d_in, const int* in_sizes, int n_in,
                              void* d_out, int out_size, void* d_ws, size_t ws_size,
                              hipStream_t stream) {
  (void)in_sizes; (void)n_in; (void)out_size; (void)ws_size;
  const float* x     = (const float*)d_in[0];
  const float* Wg    = (const float*)d_in[1];
  const float* bg    = (const float*)d_in[2];
  const float* Wth   = (const float*)d_in[3];
  const float* bth   = (const float*)d_in[4];
  const float* Wph   = (const float*)d_in[5];
  const float* bph   = (const float*)d_in[6];
  const float* Ww    = (const float*)d_in[7];
  const float* bw    = (const float*)d_in[8];
  const float* gamma = (const float*)d_in[9];
  const float* beta  = (const float*)d_in[10];
  float* out = (float*)d_out;

  char* ws = (char*)d_ws;
  size_t off = 0;
  auto alloc = [&](size_t bytes) -> void* {
    void* p = ws + off;
    off += bytes;
    off = (off + 255) & ~(size_t)255;
    return p;
  };
  unsigned short* Wall    = (unsigned short*)alloc((size_t)384 * CH * 2);
  unsigned short* WwB     = (unsigned short*)alloc((size_t)CH * ICH * 2);
  float*          bias_all= (float*)alloc((size_t)384 * 4);
  float*          chsum   = (float*)alloc((size_t)CH * 4);    // contiguous with
  float*          chsumsq = (float*)alloc((size_t)CH * 4);    // chsum (one memset)
  float*          scale   = (float*)alloc((size_t)CH * 4);
  float*          shift   = (float*)alloc((size_t)CH * 4);
  unsigned short* PROJ    = (unsigned short*)alloc((size_t)BATCH * NPIX * PROJW * 2);
  unsigned short* GT      = (unsigned short*)alloc((size_t)BATCH * ICH * NPIX * 2);
  float*          WY      = (float*)alloc((size_t)BATCH * CH * NPIX * 4);

  hipMemsetAsync(chsum, 0, 2 * CH * sizeof(float), stream);

  const int prep_n = 384 * CH + CH * ICH + 384;
  nlb_prep<<<dim3((prep_n + 255) / 256), dim3(256), 0, stream>>>(
      Wg, bg, Wth, bth, Wph, bph, Ww, Wall, WwB, bias_all);
  nlb_proj<<<dim3(BATCH * (NPIX / 16)), dim3(256), 0, stream>>>(x, Wall, bias_all, PROJ, GT);
  nlb_attn<<<dim3(BATCH * (NPIX / 16)), dim3(256), 0, stream>>>(PROJ, GT, WwB, bw, WY, chsum, chsumsq);
  nlb_stats<<<dim3(1), dim3(256), 0, stream>>>(chsum, chsumsq, gamma, beta, scale, shift);
  nlb_apply<<<dim3((BATCH * CH * NPIX) / 1024), dim3(256), 0, stream>>>(WY, x, scale, shift, out);
}
